// MultiheadAttention_70239895158997
// MI455X (gfx1250) — compile-verified, matrix operand order checked by reading, not run on hardware
//
#include <hip/hip_runtime.h>
#include <hip/hip_bf16.h>

// ---------------- CDNA5 wave32 / WMMA helpers ----------------
typedef __attribute__((ext_vector_type(16))) __bf16 v16bf;
typedef __attribute__((ext_vector_type(8)))  __bf16 bf16x8;
typedef __attribute__((ext_vector_type(8)))  float  v8f;

// gfx1250 has hw bf16 conversion; let clang lower fptrunc (RNE) itself.
static __device__ __forceinline__ __bf16 f2bf(float f) { return (__bf16)f; }

static __device__ __forceinline__ bf16x8 cvt8(float4 a, float4 b) {
  bf16x8 r;
  r[0] = (__bf16)a.x; r[1] = (__bf16)a.y; r[2] = (__bf16)a.z; r[3] = (__bf16)a.w;
  r[4] = (__bf16)b.x; r[5] = (__bf16)b.y; r[6] = (__bf16)b.z; r[7] = (__bf16)b.w;
  return r;
}

static __device__ __forceinline__ v16bf mk16(bf16x8 lo, bf16x8 hi) {
  v16bf r;
#pragma unroll
  for (int i = 0; i < 8; ++i) { r[i] = lo[i]; r[i + 8] = hi[i]; }
  return r;
}

static __device__ __forceinline__ v8f wmma_bf16(v16bf a, v16bf b, v8f c) {
  return __builtin_amdgcn_wmma_f32_16x16x32_bf16(false, a, false, b,
                                                 (short)0, c, false, false);
}

#define EMB 512
#define SEQ 4096
#define HEADS 8
#define DHEAD 64

// ================================================================
// Kernel 1: projection GEMM  Y(b,h,s,d) = X(m,:)*W(:,n) + bias[n]
// Block 256 thr = 8 waves; block tile 128(M) x 64(N); wave tile 32x32
// (2x2 WMMA tiles with operand reuse). K-steps of 32.
// ================================================================
__global__ __launch_bounds__(256) void proj_gemm(
    const float* __restrict__ X, const float* __restrict__ W,
    const float* __restrict__ bias, __bf16* __restrict__ Y) {
  __shared__ __attribute__((aligned(16))) __bf16 As[128][32];
  __shared__ __attribute__((aligned(16))) __bf16 Bt[64][32];  // W transposed: Bt[n][k]

  const int tid = threadIdx.x;
  const int lane = tid & 31, wave = tid >> 5;
  const int wy = wave >> 1, wx = wave & 1;          // wave tile origin (32x32)
  const int hi16 = lane >> 4, ln = lane & 15;
  const int n0 = blockIdx.x * 64;
  const int m0 = blockIdx.y * 128;

  const int arow = tid >> 1, acol = (tid & 1) * 16;  // A: 128x32, 16 f32/thr
  const int brow = tid >> 3, bcol = (tid & 7) * 8;   // B: 32x64,   8 f32/thr

  v8f acc[2][2] = {};
  for (int kk = 0; kk < EMB; kk += 32) {
    const float* xr = &X[(size_t)(m0 + arow) * EMB + kk + acol];
    const float4 a0 = *(const float4*)&xr[0], a1 = *(const float4*)&xr[4];
    const float4 a2 = *(const float4*)&xr[8], a3 = *(const float4*)&xr[12];
    *(bf16x8*)&As[arow][acol]     = cvt8(a0, a1);
    *(bf16x8*)&As[arow][acol + 8] = cvt8(a2, a3);

    const float* wr = &W[(size_t)(kk + brow) * EMB + n0 + bcol];
    const float4 b0 = *(const float4*)&wr[0], b1 = *(const float4*)&wr[4];
    Bt[bcol + 0][brow] = f2bf(b0.x); Bt[bcol + 1][brow] = f2bf(b0.y);
    Bt[bcol + 2][brow] = f2bf(b0.z); Bt[bcol + 3][brow] = f2bf(b0.w);
    Bt[bcol + 4][brow] = f2bf(b1.x); Bt[bcol + 5][brow] = f2bf(b1.y);
    Bt[bcol + 6][brow] = f2bf(b1.z); Bt[bcol + 7][brow] = f2bf(b1.w);
    __syncthreads();

    // per-lane k chunks: j=0..7 -> 8*(lane>=16); j=8..15 -> 16 + 8*(lane>=16)
    const int c0 = 8 * hi16, c1 = 16 + 8 * hi16;
    v16bf av[2], bv2[2];
#pragma unroll
    for (int i = 0; i < 2; ++i) {
      const int r = wy * 32 + i * 16 + ln;
      av[i] = mk16(*(const bf16x8*)&As[r][c0], *(const bf16x8*)&As[r][c1]);
      const int n = wx * 32 + i * 16 + ln;
      bv2[i] = mk16(*(const bf16x8*)&Bt[n][c0], *(const bf16x8*)&Bt[n][c1]);
    }
#pragma unroll
    for (int mi = 0; mi < 2; ++mi)
#pragma unroll
      for (int ni = 0; ni < 2; ++ni)
        acc[mi][ni] = wmma_bf16(av[mi], bv2[ni], acc[mi][ni]);
    __syncthreads();
  }

#pragma unroll
  for (int ni = 0; ni < 2; ++ni) {
    const int gn = n0 + wx * 32 + ni * 16 + ln;
    const int h = gn >> 6, d = gn & (DHEAD - 1);
    const float bvv = bias[gn];
#pragma unroll
    for (int mi = 0; mi < 2; ++mi) {
#pragma unroll
      for (int r = 0; r < 8; ++r) {
        const int gm = m0 + wy * 32 + mi * 16 + r + 8 * hi16;  // M = r + 8*(lane>=16)
        const int bb = gm >> 12, ss = gm & (SEQ - 1);
        Y[(((size_t)bb * HEADS + h) * SEQ + ss) * DHEAD + d] =
            f2bf(acc[mi][ni][r] + bvv);
      }
    }
  }
}

// ================================================================
// Kernel 2: flash attention per (b,h). Q tile 128 rows, 64-key blocks.
// 8 waves, wave w owns rows w*16..w*16+15 of the Q tile.
// ================================================================
__global__ __launch_bounds__(256) void flash_attn(
    const __bf16* __restrict__ Q, const __bf16* __restrict__ K,
    const __bf16* __restrict__ V, const int* __restrict__ mask,
    __bf16* __restrict__ ctx) {
  __shared__ __attribute__((aligned(16))) __bf16 Qs[128][64];
  __shared__ __attribute__((aligned(16))) __bf16 Ks[64][64];      // Ks[key][d]
  __shared__ __attribute__((aligned(16))) __bf16 Vt[64][64];      // Vt[d][key]
  __shared__ __attribute__((aligned(16))) __bf16 Ps[8][16][64];   // per-wave P tile

  const int tid = threadIdx.x, lane = tid & 31, w = tid >> 5;
  const int hi16 = lane >> 4, ln = lane & 15;
  const int bh = blockIdx.y, b = bh >> 3, h = bh & 7;
  const int q0 = blockIdx.x * 128;
  const size_t base = (size_t)bh * SEQ * DHEAD;
  const __bf16* Qp = Q + base + (size_t)q0 * DHEAD;
  const __bf16* Kp = K + base;
  const __bf16* Vp = V + base;
  const int* mp = mask + b * SEQ;

  {  // stage Q tile: 128x64 bf16, 64B per thread
    const int row = tid >> 1, col = (tid & 1) * 32;
    const bf16x8* src = (const bf16x8*)&Qp[(size_t)row * DHEAD + col];
    bf16x8* dst = (bf16x8*)&Qs[row][col];
    dst[0] = src[0]; dst[1] = src[1]; dst[2] = src[2]; dst[3] = src[3];
  }

  v8f o0 = {}, o1 = {}, o2 = {}, o3 = {};
  float mrow[8], lrow[8];
#pragma unroll
  for (int r = 0; r < 8; ++r) { mrow[r] = -3.0e38f; lrow[r] = 0.f; }
  const float scale = 0.125f;  // 1/sqrt(64)

  const int ldrow = tid >> 2, ldcol = (tid & 3) * 16;

  for (int kb = 0; kb < SEQ; kb += 64) {
    __syncthreads();  // prior iteration's WMMA reads done before overwrite
    {
      const bf16x8* ks = (const bf16x8*)&Kp[(size_t)(kb + ldrow) * DHEAD + ldcol];
      *(bf16x8*)&Ks[ldrow][ldcol] = ks[0];
      *(bf16x8*)&Ks[ldrow][ldcol + 8] = ks[1];
      const __bf16* vs = &Vp[(size_t)(kb + ldrow) * DHEAD + ldcol];
#pragma unroll
      for (int i = 0; i < 16; ++i) Vt[ldcol + i][ldrow] = vs[i];  // transpose V
    }
    if (kb + 64 < SEQ) {  // global_prefetch_b8 of next K/V tiles
      __builtin_prefetch(&Kp[(size_t)(kb + 64 + ldrow) * DHEAD + ldcol], 0, 1);
      __builtin_prefetch(&Vp[(size_t)(kb + 64 + ldrow) * DHEAD + ldcol], 0, 1);
    }
    __syncthreads();

    // scores S = Q Kt : 4 N-tiles x 2 K-steps = 8 WMMAs / wave
    v8f s0 = {}, s1 = {}, s2 = {}, s3 = {};
#pragma unroll
    for (int ks = 0; ks < DHEAD; ks += 32) {
      const int c0 = ks + 8 * hi16, c1 = ks + 16 + 8 * hi16;
      v16bf a = mk16(*(const bf16x8*)&Qs[w * 16 + ln][c0],
                     *(const bf16x8*)&Qs[w * 16 + ln][c1]);
      s0 = wmma_bf16(a, mk16(*(const bf16x8*)&Ks[ln][c0],
                             *(const bf16x8*)&Ks[ln][c1]), s0);
      s1 = wmma_bf16(a, mk16(*(const bf16x8*)&Ks[16 + ln][c0],
                             *(const bf16x8*)&Ks[16 + ln][c1]), s1);
      s2 = wmma_bf16(a, mk16(*(const bf16x8*)&Ks[32 + ln][c0],
                             *(const bf16x8*)&Ks[32 + ln][c1]), s2);
      s3 = wmma_bf16(a, mk16(*(const bf16x8*)&Ks[48 + ln][c0],
                             *(const bf16x8*)&Ks[48 + ln][c1]), s3);
    }

    // online softmax: row M=r+8*hi16 lives across one 16-lane half
    const int m0v = (mp[kb + ln] == 0), m1v = (mp[kb + 16 + ln] == 0);
    const int m2v = (mp[kb + 32 + ln] == 0), m3v = (mp[kb + 48 + ln] == 0);
#pragma unroll
    for (int r = 0; r < 8; ++r) {
      float v0 = m0v ? -1e9f : s0[r] * scale;
      float v1 = m1v ? -1e9f : s1[r] * scale;
      float v2 = m2v ? -1e9f : s2[r] * scale;
      float v3 = m3v ? -1e9f : s3[r] * scale;
      float mx = fmaxf(fmaxf(v0, v1), fmaxf(v2, v3));
#pragma unroll
      for (int m = 1; m < 16; m <<= 1) mx = fmaxf(mx, __shfl_xor(mx, m, 32));
      const float mnew = fmaxf(mrow[r], mx);
      const float alpha = __expf(mrow[r] - mnew);
      const float p0 = __expf(v0 - mnew), p1 = __expf(v1 - mnew);
      const float p2 = __expf(v2 - mnew), p3 = __expf(v3 - mnew);
      float rs = p0 + p1 + p2 + p3;
#pragma unroll
      for (int m = 1; m < 16; m <<= 1) rs += __shfl_xor(rs, m, 32);
      lrow[r] = lrow[r] * alpha + rs;
      mrow[r] = mnew;
      o0[r] = o0[r] * alpha; o1[r] = o1[r] * alpha;
      o2[r] = o2[r] * alpha; o3[r] = o3[r] * alpha;
      const int prow = r + 8 * hi16;
      Ps[w][prow][ln]      = f2bf(p0);
      Ps[w][prow][16 + ln] = f2bf(p1);
      Ps[w][prow][32 + ln] = f2bf(p2);
      Ps[w][prow][48 + ln] = f2bf(p3);
    }

    // O += P(16x64) x V(64x64): same-wave LDS RAW, DScnt-ordered
#pragma unroll
    for (int ks = 0; ks < 64; ks += 32) {
      const int c0 = ks + 8 * hi16, c1 = ks + 16 + 8 * hi16;
      v16bf a = mk16(*(const bf16x8*)&Ps[w][ln][c0],
                     *(const bf16x8*)&Ps[w][ln][c1]);
      o0 = wmma_bf16(a, mk16(*(const bf16x8*)&Vt[ln][c0],
                             *(const bf16x8*)&Vt[ln][c1]), o0);
      o1 = wmma_bf16(a, mk16(*(const bf16x8*)&Vt[16 + ln][c0],
                             *(const bf16x8*)&Vt[16 + ln][c1]), o1);
      o2 = wmma_bf16(a, mk16(*(const bf16x8*)&Vt[32 + ln][c0],
                             *(const bf16x8*)&Vt[32 + ln][c1]), o2);
      o3 = wmma_bf16(a, mk16(*(const bf16x8*)&Vt[48 + ln][c0],
                             *(const bf16x8*)&Vt[48 + ln][c1]), o3);
    }
  }

#pragma unroll
  for (int r = 0; r < 8; ++r) {
    const float inv = 1.f / lrow[r];
    const int row = q0 + w * 16 + r + 8 * hi16;
    const size_t ob = ((size_t)b * SEQ + row) * EMB + h * DHEAD;
    ctx[ob + ln]      = f2bf(o0[r] * inv);
    ctx[ob + 16 + ln] = f2bf(o1[r] * inv);
    ctx[ob + 32 + ln] = f2bf(o2[r] * inv);
    ctx[ob + 48 + ln] = f2bf(o3[r] * inv);
  }
}

// ================================================================
// Kernel 3: output GEMM  out(m,n) = ctx_bf16(m,:)*W_o(:,n) + b_o[n], f32 out
// Same 128x64 block tile / 32x32 wave tile as proj_gemm.
// ================================================================
__global__ __launch_bounds__(256) void out_gemm(
    const __bf16* __restrict__ A, const float* __restrict__ W,
    const float* __restrict__ bias, float* __restrict__ out) {
  __shared__ __attribute__((aligned(16))) __bf16 As[128][32];
  __shared__ __attribute__((aligned(16))) __bf16 Bt[64][32];

  const int tid = threadIdx.x;
  const int lane = tid & 31, wave = tid >> 5;
  const int wy = wave >> 1, wx = wave & 1;
  const int hi16 = lane >> 4, ln = lane & 15;
  const int n0 = blockIdx.x * 64;
  const int m0 = blockIdx.y * 128;
  const int arow = tid >> 1, acol = (tid & 1) * 16;
  const int brow = tid >> 3, bcol = (tid & 7) * 8;

  v8f acc[2][2] = {};
  for (int kk = 0; kk < EMB; kk += 32) {
    const bf16x8* src = (const bf16x8*)&A[(size_t)(m0 + arow) * EMB + kk + acol];
    bf16x8* dst = (bf16x8*)&As[arow][acol];
    dst[0] = src[0]; dst[1] = src[1];

    const float* wr = &W[(size_t)(kk + brow) * EMB + n0 + bcol];
    const float4 b0 = *(const float4*)&wr[0], b1 = *(const float4*)&wr[4];
    Bt[bcol + 0][brow] = f2bf(b0.x); Bt[bcol + 1][brow] = f2bf(b0.y);
    Bt[bcol + 2][brow] = f2bf(b0.z); Bt[bcol + 3][brow] = f2bf(b0.w);
    Bt[bcol + 4][brow] = f2bf(b1.x); Bt[bcol + 5][brow] = f2bf(b1.y);
    Bt[bcol + 6][brow] = f2bf(b1.z); Bt[bcol + 7][brow] = f2bf(b1.w);
    __syncthreads();

    const int c0 = 8 * hi16, c1 = 16 + 8 * hi16;
    v16bf av[2], bv2[2];
#pragma unroll
    for (int i = 0; i < 2; ++i) {
      const int r = wy * 32 + i * 16 + ln;
      av[i] = mk16(*(const bf16x8*)&As[r][c0], *(const bf16x8*)&As[r][c1]);
      const int n = wx * 32 + i * 16 + ln;
      bv2[i] = mk16(*(const bf16x8*)&Bt[n][c0], *(const bf16x8*)&Bt[n][c1]);
    }
#pragma unroll
    for (int mi = 0; mi < 2; ++mi)
#pragma unroll
      for (int ni = 0; ni < 2; ++ni)
        acc[mi][ni] = wmma_bf16(av[mi], bv2[ni], acc[mi][ni]);
    __syncthreads();
  }

#pragma unroll
  for (int ni = 0; ni < 2; ++ni) {
    const int gn = n0 + wx * 32 + ni * 16 + ln;
    const float bvv = bias[gn];
#pragma unroll
    for (int mi = 0; mi < 2; ++mi) {
#pragma unroll
      for (int r = 0; r < 8; ++r) {
        const int gm = m0 + wy * 32 + mi * 16 + r + 8 * hi16;
        out[(size_t)gm * EMB + gn] = acc[mi][ni][r] + bvv;
      }
    }
  }
}

// ================================================================
extern "C" void kernel_launch(void* const* d_in, const int* in_sizes, int n_in,
                              void* d_out, int out_size, void* d_ws, size_t ws_size,
                              hipStream_t stream) {
  const float* q    = (const float*)d_in[0];
  const float* k    = (const float*)d_in[1];
  const float* v    = (const float*)d_in[2];
  const int*   mask = (const int*)d_in[3];
  const float* Wq = (const float*)d_in[4];  const float* bq = (const float*)d_in[5];
  const float* Wk = (const float*)d_in[6];  const float* bk = (const float*)d_in[7];
  const float* Wv = (const float*)d_in[8];  const float* bv = (const float*)d_in[9];
  const float* Wo = (const float*)d_in[10]; const float* bo = (const float*)d_in[11];

  const size_t NTOK = (size_t)2 * SEQ * EMB;  // 4,194,304 elements
  __bf16* Qbf = (__bf16*)d_ws;
  __bf16* Kbf = Qbf + NTOK;
  __bf16* Vbf = Kbf + NTOK;
  __bf16* ctx = Vbf + NTOK;   // total 32 MB of workspace

  dim3 blk(256);
  dim3 gemm_grid(EMB / 64, (2 * SEQ) / 128);  // 8 x 64

  proj_gemm<<<gemm_grid, blk, 0, stream>>>(q, Wq, bq, Qbf);
  proj_gemm<<<gemm_grid, blk, 0, stream>>>(k, Wk, bk, Kbf);
  proj_gemm<<<gemm_grid, blk, 0, stream>>>(v, Wv, bv, Vbf);
  flash_attn<<<dim3(SEQ / 128, 2 * HEADS), blk, 0, stream>>>(Qbf, Kbf, Vbf, mask, ctx);
  out_gemm<<<gemm_grid, blk, 0, stream>>>(ctx, Wo, bo, (float*)d_out);
}
